// DARNN_64553358459246
// MI455X (gfx1250) — compile-verified, matrix operand order checked by reading, not run on hardware
//
#include <hip/hip_runtime.h>
#include <hip/hip_bf16.h>

// ---------------------------------------------------------------------------
// DA-RNN forward for MI455X (gfx1250, wave32, WMMA 16x16x32 f16).
// B=1024, T=128, N=128, M=P=256.
// Persistent recurrent kernels: 256 blocks x 4 batch rows (tanh/trans VALU is
// the bottleneck -> maximize WGP coverage; WMMA runs 16-row tiles with rows
// 4..15 held at zero).  Big tensors (UeD 33.5MB, H 67MB, UH 67MB) kept f16 so
// the working set (~168MB) is L2-resident (192MB L2).
// ---------------------------------------------------------------------------

typedef _Float16 h16;
typedef __attribute__((ext_vector_type(16))) _Float16 v16h;
typedef __attribute__((ext_vector_type(8)))  _Float16 v8h;
typedef __attribute__((ext_vector_type(8)))  float    v8f;

#define L2E 1.4426950408889634f
#define BR  4                      // batch rows per persistent block

__device__ __forceinline__ float fast_exp(float x) {
    return __builtin_amdgcn_exp2f(x * L2E);
}
__device__ __forceinline__ float fast_sigmoid(float x) {
    x = fminf(fmaxf(x, -30.f), 30.f);
    return __builtin_amdgcn_rcpf(1.f + __builtin_amdgcn_exp2f(-x * L2E));
}
__device__ __forceinline__ float fast_tanh(float x) {
    x = fminf(fmaxf(x, -15.f), 15.f);
    float t = __builtin_amdgcn_exp2f(-2.f * L2E * x);
    return (1.f - t) * __builtin_amdgcn_rcpf(1.f + t);
}

// ---- WMMA fragment helpers (ISA 7.12.2 layouts, wave32) -------------------
// A 16x32 f16: lane holds row M=lane&15; K base (lane>>4)*8; VGPR i covers
// K = (i>=4?16:0) + kbase + 2*(i&3).  B (as W^T, W row-major [n_out x K])
// mirrors it: lane holds column N=lane&15, same K map -> contiguous pairs.
__device__ __forceinline__ v16h load_frag(const h16* p, int ld) {
    int lane = threadIdx.x & 31;
    const h16* q = p + (lane & 15) * ld + ((lane >> 4) << 3);
    v16h f;
#pragma unroll
    for (int i = 0; i < 8; ++i) {
        int k = ((i & 4) << 2) + ((i & 3) << 1);
        f[2 * i]     = q[k];
        f[2 * i + 1] = q[k + 1];
    }
    return f;
}

__device__ __forceinline__ v8f wmma_f16(v16h a, v16h b, v8f c) {
    return __builtin_amdgcn_wmma_f32_16x16x32_f16(false, a, false, b,
                                                  (short)0, c, false, false);
}

// D/C 16x16 f32: VGPR r, lanes 0-15 -> (M=r, N=lane); lanes 16-31 -> (M=r+8).
__device__ __forceinline__ void store_acc_f32(float* out, int ld, v8f c) {
    int lane = threadIdx.x & 31;
    int n  = lane & 15;
    int mb = (lane >> 4) << 3;
#pragma unroll
    for (int r = 0; r < 8; ++r) out[(mb + r) * ld + n] = c[r];
}
__device__ __forceinline__ void store_acc_h16(h16* out, int ld, v8f c) {
    int lane = threadIdx.x & 31;
    int n  = lane & 15;
    int mb = (lane >> 4) << 3;
#pragma unroll
    for (int r = 0; r < 8; ++r) out[(mb + r) * ld + n] = (h16)c[r];
}

// ---------------------------------------------------------------------------
__global__ void cvt_kernel(h16* __restrict__ dst, const float* __restrict__ src, int n) {
    int i = blockIdx.x * blockDim.x + threadIdx.x;
    if (i < n) dst[i] = (h16)src[i];
}

// ---------------------------------------------------------------------------
// UeD[b,n,s] = sum_t enc_data[b,t,n] * Ue[s,t]   (per-b 128x128 = Xb^T @ Ue^T)
__global__ __launch_bounds__(256) void ue_data_kernel(
    const float* __restrict__ enc_data, const h16* __restrict__ Ue_h,
    h16* __restrict__ UeD) {
    __shared__ h16 xT[128][136];               // [n][t], padded
    int b = blockIdx.x;
    const float* src = enc_data + (long)b * (128 * 128);
    for (int i = threadIdx.x; i < 128 * 128; i += 256) {
        int t = i >> 7, n = i & 127;
        xT[n][t] = (h16)src[i];
    }
    __syncthreads();
    int wave = threadIdx.x >> 5;
    h16* dst = UeD + (long)b * (128 * 128);
    for (int tt = wave; tt < 64; tt += 8) {    // 8x8 output tiles of 16x16
        int r0 = (tt >> 3) << 4, c0 = (tt & 7) << 4;
        v8f acc = {};
#pragma unroll
        for (int kt = 0; kt < 4; ++kt) {
            v16h a  = load_frag(&xT[r0][kt * 32], 136);
            v16h bf = load_frag(Ue_h + c0 * 128 + kt * 32, 128);
            acc = wmma_f16(a, bf, acc);
        }
        store_acc_h16(dst + r0 * 128 + c0, 128, acc);
    }
}

// ---------------------------------------------------------------------------
// UH[bt,u] = sum_m H[bt,m] * Ud[u,m]   (131072 x 256 GEMM, K=256)
// A tile (16x256 of H) staged into LDS with async global->LDS copies
// (ASYNCcnt-tracked CDNA5 path), fragments then built from LDS.
__global__ __launch_bounds__(256) void uh_kernel(
    const h16* __restrict__ H, const h16* __restrict__ Ud_h, h16* __restrict__ UH) {
    __shared__ h16 hbuf[16 * 256];             // 8KB A tile
    long row0 = (long)blockIdx.x * 16;
    int tid = threadIdx.x, wave = tid >> 5;

    unsigned ldsbase = (unsigned)(size_t)&hbuf[0];
    for (int j = tid; j < 512; j += 256) {     // 512 x 16B transfers
        int row = j >> 5, seg = j & 31;
        unsigned long long gaddr =
            (unsigned long long)(size_t)(H + (row0 + row) * 256 + seg * 8);
        unsigned laddr = ldsbase + (unsigned)((row * 256 + seg * 8) * 2);
        asm volatile("global_load_async_to_lds_b128 %0, %1, off"
                     :: "v"(laddr), "v"(gaddr) : "memory");
    }
    asm volatile("s_wait_asynccnt 0x0" ::: "memory");
    __syncthreads();

    v16h af[8];
#pragma unroll
    for (int kt = 0; kt < 8; ++kt) af[kt] = load_frag(&hbuf[kt * 32], 256);
    for (int ct = wave; ct < 16; ct += 8) {
        v8f acc = {};
#pragma unroll
        for (int kt = 0; kt < 8; ++kt)
            acc = wmma_f16(af[kt], load_frag(Ud_h + ct * 16 * 256 + kt * 32, 256), acc);
        store_acc_h16(UH + row0 * 256 + ct * 16, 256, acc);
    }
}

// ---------------------------------------------------------------------------
// Encoder: persistent, 256 blocks x BR=4 batch rows, 128 recurrent steps.
__global__ __launch_bounds__(512) void enc_kernel(
    const float* __restrict__ enc_data, const float* __restrict__ bih,
    const float* __restrict__ bhh, const float* __restrict__ ve,
    const h16* __restrict__ Wih, const h16* __restrict__ Whh,
    const h16* __restrict__ We, const h16* __restrict__ UeD,
    h16* __restrict__ H) {
    __shared__ h16   q_l[16][512];   // [h f16 | c f16]; rows BR..15 stay zero
    __shared__ float c_l[BR][256];
    __shared__ float wq_l[16][128];
    __shared__ float sc_l[BR][128];
    __shared__ h16   xt_l[16][128];  // rows BR..15 stay zero
    __shared__ float g_l[16][1024];
    __shared__ float ve_l[128];

    int tid = threadIdx.x, wave = tid >> 5, lane = tid & 31;
    int b0 = blockIdx.x * BR;

    for (int i = tid; i < 16 * 512; i += 512) (&q_l[0][0])[i] = (h16)0.f;
    for (int i = tid; i < 16 * 128; i += 512) (&xt_l[0][0])[i] = (h16)0.f;
    for (int i = tid; i < BR * 256; i += 512) (&c_l[0][0])[i] = 0.f;
    if (tid < 128) ve_l[tid] = ve[tid];
    __syncthreads();

    for (int t = 0; t < 128; ++t) {
        // ---- wq = [h;c] @ We^T  (16x512 x 512x128) ----
        if (wave < 8) {
            int s0 = wave << 4;
            v8f acc = {};
#pragma unroll
            for (int kt = 0; kt < 16; ++kt) {
                v16h a = load_frag(&q_l[0][kt * 32], 512);
                v16h b = load_frag(We + s0 * 512 + kt * 32, 512);
                acc = wmma_f16(a, b, acc);
            }
            store_acc_f32(&wq_l[0][s0], 128, acc);
        }
        __syncthreads();
        // ---- score[b,n] = sum_s tanh(wq[b,s] + UeD[b,n,s]) * ve[s] ----
        {   // BR*128 = 512 rows, one per thread; vectorized 16B f16 loads
            int bi = tid >> 7, n = tid & 127;
            const v8h* ur = (const v8h*)(UeD + (((long)(b0 + bi)) * 128 + n) * 128);
            float s = 0.f;
#pragma unroll 4
            for (int v = 0; v < 16; ++v) {
                v8h hv = ur[v];
#pragma unroll
                for (int e = 0; e < 8; ++e) {
                    int k = v * 8 + e;
                    s += fast_tanh(wq_l[bi][k] + (float)hv[e]) * ve_l[k];
                }
            }
            sc_l[bi][n] = s;
        }
        __syncthreads();
        // ---- softmax over n, scale x_t ----
        if (tid < BR * 16) {
            int bi = tid >> 4, sub = tid & 15;
            float m = -1e30f;
            for (int n = sub; n < 128; n += 16) m = fmaxf(m, sc_l[bi][n]);
#pragma unroll
            for (int o = 8; o > 0; o >>= 1) m = fmaxf(m, __shfl_xor(m, o, 16));
            float sum = 0.f;
            for (int n = sub; n < 128; n += 16) sum += fast_exp(sc_l[bi][n] - m);
#pragma unroll
            for (int o = 8; o > 0; o >>= 1) sum += __shfl_xor(sum, o, 16);
            float inv = __builtin_amdgcn_rcpf(sum);
            const float* xr = enc_data + ((long)(b0 + bi) * 128 + t) * 128;
            for (int n = sub; n < 128; n += 16) {
                float al = fast_exp(sc_l[bi][n] - m) * inv;
                xt_l[bi][n] = (h16)(al * xr[n]);
            }
        }
        __syncthreads();
        // ---- gates = xt @ Wih^T + h @ Whh^T + bih + bhh ----
        {
            v16h ax[4], ah[8];
#pragma unroll
            for (int kt = 0; kt < 4; ++kt) ax[kt] = load_frag(&xt_l[0][kt * 32], 128);
#pragma unroll
            for (int kt = 0; kt < 8; ++kt) ah[kt] = load_frag(&q_l[0][kt * 32], 512);
            for (int nt = wave; nt < 64; nt += 16) {
                int n0 = nt << 4;
                v8f acc = {};
#pragma unroll
                for (int kt = 0; kt < 4; ++kt)
                    acc = wmma_f16(ax[kt], load_frag(Wih + n0 * 128 + kt * 32, 128), acc);
#pragma unroll
                for (int kt = 0; kt < 8; ++kt)
                    acc = wmma_f16(ah[kt], load_frag(Whh + n0 * 256 + kt * 32, 256), acc);
                int cn = n0 + (lane & 15);
                float bsum = bih[cn] + bhh[cn];
                int mb = (lane >> 4) << 3;
#pragma unroll
                for (int r = 0; r < 8; ++r) g_l[mb + r][cn] = acc[r] + bsum;
            }
        }
        __syncthreads();
        // ---- LSTM state update, emit h ----
        for (int idx = tid; idx < BR * 256; idx += 512) {
            int bi = idx >> 8, j = idx & 255;
            float gi = g_l[bi][j],       gf = g_l[bi][j + 256];
            float gg = g_l[bi][j + 512], go = g_l[bi][j + 768];
            float c = fast_sigmoid(gf) * c_l[bi][j] + fast_sigmoid(gi) * fast_tanh(gg);
            float h = fast_sigmoid(go) * fast_tanh(c);
            c_l[bi][j] = c;
            q_l[bi][j] = (h16)h;
            q_l[bi][256 + j] = (h16)c;
            H[(((long)(b0 + bi)) * 128 + t) * 256 + j] = (h16)h;
        }
        __syncthreads();
    }
}

// ---------------------------------------------------------------------------
// Decoder: persistent, 256 blocks x BR rows, 127 steps + final attn + MLP.
__global__ __launch_bounds__(512) void dec_kernel(
    const float* __restrict__ dec_data, const float* __restrict__ vd,
    const float* __restrict__ wt_w, const float* __restrict__ wt_b,
    const float* __restrict__ dWih, const float* __restrict__ bih,
    const float* __restrict__ bhh, const float* __restrict__ l1_w,
    const float* __restrict__ l1_b, const float* __restrict__ l2_w,
    const float* __restrict__ l2_b, const h16* __restrict__ Wd,
    const h16* __restrict__ dWhh, const h16* __restrict__ H,
    const h16* __restrict__ UH, float* __restrict__ out) {
    __shared__ h16   q_l[16][512];   // [d f16 | s f16]; rows BR..15 stay zero
    __shared__ float s_l[BR][256];
    __shared__ float wq_l[16][256];
    __shared__ float sc_l[BR][128];
    __shared__ float ctx_l[BR][256];
    __shared__ float g_l[16][1024];
    __shared__ float vd_l[256];
    __shared__ float yt_l[16];

    int tid = threadIdx.x, wave = tid >> 5, lane = tid & 31;
    int b0 = blockIdx.x * BR;

    for (int i = tid; i < 16 * 512; i += 512) (&q_l[0][0])[i] = (h16)0.f;
    for (int i = tid; i < BR * 256; i += 512) (&s_l[0][0])[i] = 0.f;
    if (tid < 256) vd_l[tid] = vd[tid];
    if (tid < 16)  yt_l[tid] = 0.f;
    __syncthreads();

    for (int step = 0; step < 128; ++step) {
        // ======== temporal attention: ctx = softmax(tanh(q@Wd^T + UH)@vd) @ H
        {   // wq = [d;s] @ Wd^T  (16x512 x 512x256): one 16-col tile per wave
            int u0 = wave << 4;
            v8f acc = {};
#pragma unroll
            for (int kt = 0; kt < 16; ++kt) {
                v16h a = load_frag(&q_l[0][kt * 32], 512);
                v16h b = load_frag(Wd + u0 * 512 + kt * 32, 512);
                acc = wmma_f16(a, b, acc);
            }
            store_acc_f32(&wq_l[0][u0], 256, acc);
        }
        __syncthreads();
        {   // score[b,tt]: BR*128 = 512 rows, one per thread, 16B loads
            int bi = tid >> 7, tt = tid & 127;
            const v8h* ur = (const v8h*)(UH + (((long)(b0 + bi)) * 128 + tt) * 256);
            float s = 0.f;
#pragma unroll 4
            for (int v = 0; v < 32; ++v) {
                v8h hv = ur[v];
#pragma unroll
                for (int e = 0; e < 8; ++e) {
                    int u = v * 8 + e;
                    s += fast_tanh(wq_l[bi][u] + (float)hv[e]) * vd_l[u];
                }
            }
            sc_l[bi][tt] = s;
        }
        __syncthreads();
        if (tid < BR * 16) {                            // softmax over tt
            int bi = tid >> 4, sub = tid & 15;
            float m = -1e30f;
            for (int n = sub; n < 128; n += 16) m = fmaxf(m, sc_l[bi][n]);
#pragma unroll
            for (int o = 8; o > 0; o >>= 1) m = fmaxf(m, __shfl_xor(m, o, 16));
            float sum = 0.f;
            for (int n = sub; n < 128; n += 16) sum += fast_exp(sc_l[bi][n] - m);
#pragma unroll
            for (int o = 8; o > 0; o >>= 1) sum += __shfl_xor(sum, o, 16);
            float inv = __builtin_amdgcn_rcpf(sum);
            for (int n = sub; n < 128; n += 16)
                sc_l[bi][n] = fast_exp(sc_l[bi][n] - m) * inv;  // beta
        }
        // zero ctx accumulators
        for (int i = tid; i < BR * 256; i += 512) (&ctx_l[0][0])[i] = 0.f;
        __syncthreads();
        {   // ctx[b,m] = sum_tt beta[b,tt] * H[b,tt,m]
            // thread = (bi, tt-quarter, m-chunk of 8): coalesced v8h rows,
            // partials merged with LDS float atomics (ds_add_f32).
            int bi = tid >> 7, rem = tid & 127;
            int tg = rem >> 5, mc = rem & 31;
            int m0 = mc << 3;
            const h16* hb = H + ((long)(b0 + bi)) * 128 * 256 + m0;
            float acc[8] = {};
#pragma unroll 4
            for (int tt = tg * 32; tt < tg * 32 + 32; ++tt) {
                float beta = sc_l[bi][tt];
                v8h hv = *(const v8h*)(hb + tt * 256);
#pragma unroll
                for (int e = 0; e < 8; ++e) acc[e] += beta * (float)hv[e];
            }
#pragma unroll
            for (int e = 0; e < 8; ++e) atomicAdd(&ctx_l[bi][m0 + e], acc[e]);
        }
        __syncthreads();
        if (step == 127) break;   // last iteration: attention only

        // ---- y_tilde = [y, ctx] @ wt_w^T + wt_b ----
        if (tid < BR * 16) {
            int bi = tid >> 4, sub = tid & 15;
            float p = 0.f;
            for (int m = sub; m < 256; m += 16) p += wt_w[1 + m] * ctx_l[bi][m];
#pragma unroll
            for (int o = 8; o > 0; o >>= 1) p += __shfl_xor(p, o, 16);
            if (sub == 0)
                yt_l[bi] = wt_b[0] + wt_w[0] * dec_data[(long)(b0 + bi) * 128 + step] + p;
        }
        __syncthreads();
        // ---- gates = y_tilde*dWih + d @ dWhh^T + bih + bhh ----
        {
            v16h ah[8];
#pragma unroll
            for (int kt = 0; kt < 8; ++kt) ah[kt] = load_frag(&q_l[0][kt * 32], 512);
            for (int nt = wave; nt < 64; nt += 16) {
                int n0 = nt << 4;
                v8f acc = {};
#pragma unroll
                for (int kt = 0; kt < 8; ++kt)
                    acc = wmma_f16(ah[kt], load_frag(dWhh + n0 * 256 + kt * 32, 256), acc);
                int cn = n0 + (lane & 15);
                float w_in = dWih[cn];
                float bsum = bih[cn] + bhh[cn];
                int mb = (lane >> 4) << 3;
#pragma unroll
                for (int r = 0; r < 8; ++r)
                    g_l[mb + r][cn] = acc[r] + yt_l[mb + r] * w_in + bsum;
            }
        }
        __syncthreads();
        // ---- LSTM state update ----
        for (int idx = tid; idx < BR * 256; idx += 512) {
            int bi = idx >> 8, j = idx & 255;
            float gi = g_l[bi][j],       gf = g_l[bi][j + 256];
            float gg = g_l[bi][j + 512], go = g_l[bi][j + 768];
            float c = fast_sigmoid(gf) * s_l[bi][j] + fast_sigmoid(gi) * fast_tanh(gg);
            float d = fast_sigmoid(go) * fast_tanh(c);
            s_l[bi][j] = c;
            q_l[bi][j] = (h16)d;
            q_l[bi][256 + j] = (h16)c;
        }
        __syncthreads();
    }

    // ======== head: out = relu([d, ctx] @ l1^T + b1) @ l2^T + b2 ========
    for (int idx = tid; idx < BR * 512; idx += 512) {   // z into g_l
        int bi = idx >> 9, j = idx & 511;
        g_l[bi][j] = (j < 256) ? (float)q_l[bi][j] : ctx_l[bi][j - 256];
    }
    __syncthreads();
    for (int idx = tid; idx < BR * 256; idx += 512) {   // hidden (relu)
        int bi = idx >> 8, j = idx & 255;
        const float* wr = l1_w + (long)j * 512;
        float s = l1_b[j];
#pragma unroll 4
        for (int k = 0; k < 512; ++k) s += wr[k] * g_l[bi][k];
        ctx_l[bi][j] = fmaxf(s, 0.f);
    }
    __syncthreads();
    if (tid < BR * 16) {
        int bi = tid >> 4, sub = tid & 15;
        float p = 0.f;
        for (int j = sub; j < 256; j += 16) p += l2_w[j] * ctx_l[bi][j];
#pragma unroll
        for (int o = 8; o > 0; o >>= 1) p += __shfl_xor(p, o, 16);
        if (sub == 0) out[b0 + bi] = l2_b[0] + p;
    }
}

// ---------------------------------------------------------------------------
extern "C" void kernel_launch(void* const* d_in, const int* in_sizes, int n_in,
                              void* d_out, int out_size, void* d_ws, size_t ws_size,
                              hipStream_t stream) {
    (void)in_sizes; (void)n_in; (void)out_size; (void)ws_size;
    const float* enc_data = (const float*)d_in[0];
    const float* dec_data = (const float*)d_in[1];
    const float* enc_Wih  = (const float*)d_in[2];
    const float* enc_Whh  = (const float*)d_in[3];
    const float* enc_bih  = (const float*)d_in[4];
    const float* enc_bhh  = (const float*)d_in[5];
    const float* We       = (const float*)d_in[6];
    const float* Ue       = (const float*)d_in[7];
    const float* ve       = (const float*)d_in[8];
    const float* Wd       = (const float*)d_in[9];
    const float* Ud       = (const float*)d_in[10];
    const float* vd       = (const float*)d_in[11];
    const float* wt_w     = (const float*)d_in[12];
    const float* wt_b     = (const float*)d_in[13];
    const float* dec_Wih  = (const float*)d_in[14];
    const float* dec_Whh  = (const float*)d_in[15];
    const float* dec_bih  = (const float*)d_in[16];
    const float* dec_bhh  = (const float*)d_in[17];
    const float* l1_w     = (const float*)d_in[18];
    const float* l1_b     = (const float*)d_in[19];
    const float* l2_w     = (const float*)d_in[20];
    const float* l2_b     = (const float*)d_in[21];
    float* out = (float*)d_out;

    // workspace layout (bytes)
    char* ws = (char*)d_ws;
    h16* H_h    = (h16*)(ws + 0);            //  67,108,864  B*T*M
    h16* UH_h   = (h16*)(ws + 67108864);     //  67,108,864  B*T*M
    h16* UeD_h  = (h16*)(ws + 134217728);    //  33,554,432  B*N*T
    h16* Wih_h  = (h16*)(ws + 167772160);    //     262,144
    h16* Whh_h  = (h16*)(ws + 168034304);    //     524,288
    h16* We_h   = (h16*)(ws + 168558592);    //     131,072
    h16* Wd_h   = (h16*)(ws + 168689664);    //     262,144
    h16* Ud_h   = (h16*)(ws + 168951808);    //     131,072
    h16* dWhh_h = (h16*)(ws + 169082880);    //     524,288
    h16* Ue_h   = (h16*)(ws + 169607168);    //      32,768

    auto cvt = [&](h16* dst, const float* src, int n) {
        cvt_kernel<<<(n + 255) / 256, 256, 0, stream>>>(dst, src, n);
    };
    cvt(Wih_h,  enc_Wih, 1024 * 128);
    cvt(Whh_h,  enc_Whh, 1024 * 256);
    cvt(We_h,   We,      128 * 512);
    cvt(Wd_h,   Wd,      256 * 512);
    cvt(Ud_h,   Ud,      256 * 256);
    cvt(dWhh_h, dec_Whh, 1024 * 256);
    cvt(Ue_h,   Ue,      128 * 128);

    ue_data_kernel<<<1024, 256, 0, stream>>>(enc_data, Ue_h, UeD_h);

    enc_kernel<<<1024 / BR, 512, 0, stream>>>(enc_data, enc_bih, enc_bhh, ve,
                                              Wih_h, Whh_h, We_h, UeD_h, H_h);

    uh_kernel<<<8192, 256, 0, stream>>>(H_h, Ud_h, UH_h);

    dec_kernel<<<1024 / BR, 512, 0, stream>>>(dec_data, vd, wt_w, wt_b, dec_Wih,
                                              dec_bih, dec_bhh, l1_w, l1_b,
                                              l2_w, l2_b, Wd_h, dWhh_h, H_h,
                                              UH_h, out);
}
